// TemporalLinear_43765716746849
// MI455X (gfx1250) — matrix operand order repaired and checked by reading, not compiled, not run on hardware
//
#include <hip/hip_runtime.h>
#include <hip/hip_bf16.h>

// Problem constants
#define T_DIM 8192
#define K_DIM 4096
#define N_DIM 4096
#define KTILES (K_DIM / 32)   // 128
#define MTILES (T_DIM / 16)   // 512
#define NTILES (N_DIM / 16)   // 256

typedef __attribute__((ext_vector_type(16))) __bf16 v16bf;
typedef __attribute__((ext_vector_type(8)))  float  v8f;

// ---------------------------------------------------------------------------
// Pack x[T,K] (fp32, row-major) into WMMA A-fragment-native tiled bf16 planes.
// A tile = 16(M) x 32(K) bf16, stored as 32 lanes x 16 contiguous bf16.
// ISA A layout (16-bit A 16x32): lane l: M = l&15; VGPR v holds 2 elems:
//   K = 16*(v>=4) + 2*(v&3) + 8*(l>=16) + pos
// Tile flat index = mt*KTILES + kt.
// ---------------------------------------------------------------------------
template <bool SPLIT>
__global__ void pack_x_kernel(const float* __restrict__ x,
                              __bf16* __restrict__ hi,
                              __bf16* __restrict__ lo) {
    int f = blockIdx.x * 256 + threadIdx.x;          // one bf16 element
    int tile = f >> 9;                               // 512 elems per tile
    int r    = f & 511;
    int lane = r >> 4;
    int j    = r & 15;
    int v    = j >> 1;
    int pos  = j & 1;
    int tm = tile >> 7;                              // / KTILES
    int tk = tile & 127;                             // % KTILES
    int m   = tm * 16 + (lane & 15);
    int kin = ((v & 4) << 2) | ((v & 3) << 1) | ((lane >> 4) << 3) | pos;
    int k   = tk * 32 + kin;
    float val = x[(size_t)m * K_DIM + k];
    __bf16 h = (__bf16)val;
    hi[f] = h;
    if (SPLIT) lo[f] = (__bf16)(val - (float)h);
}

// ---------------------------------------------------------------------------
// Pack W[N,K] (fp32, row-major) into WMMA B-fragment-native tiled bf16 planes.
// B tile = 32(K) x 16(N) bf16: lane l: N = l&15; K = 16*(j>=8) + 8*(l>=16) + (j&7).
// Tile flat index = nt*KTILES + kt (K contiguous within an N column strip).
// ---------------------------------------------------------------------------
template <bool SPLIT>
__global__ void pack_w_kernel(const float* __restrict__ W,
                              __bf16* __restrict__ hi,
                              __bf16* __restrict__ lo) {
    int f = blockIdx.x * 256 + threadIdx.x;
    int tile = f >> 9;
    int r    = f & 511;
    int lane = r >> 4;
    int j    = r & 15;
    int tn = tile >> 7;                              // / KTILES
    int tk = tile & 127;
    int n   = tn * 16 + (lane & 15);
    int kin = ((j & 8) << 1) | ((lane >> 4) << 3) | (j & 7);
    int k   = tk * 32 + kin;
    float val = W[(size_t)n * K_DIM + k];
    __bf16 h = (__bf16)val;
    hi[f] = h;
    if (SPLIT) lo[f] = (__bf16)(val - (float)h);
}

// ---------------------------------------------------------------------------
// WMMA GEMM: 256 threads = 8 waves. Block tile 128(M) x 256(N).
// Wave grid 2(M) x 4(N); each wave: 4 Mtiles x 4 Ntiles = 64x64 output,
// 16 f32 accumulators of v8f (128 VGPRs). K loop: 128 steps of depth 32.
// SPLIT: out += Ah*Bh + Ah*Bl + Al*Bh  (bf16x3, ~fp32 accuracy).
// 48 WMMAs per 16 fragment loads per K-step (1.5 WMMA per b128 load).
// ---------------------------------------------------------------------------
template <bool SPLIT>
__global__ void __launch_bounds__(256)
gemm_wmma_kernel(const __bf16* __restrict__ Ah, const __bf16* __restrict__ Al,
                 const __bf16* __restrict__ Bh, const __bf16* __restrict__ Bl,
                 const float* __restrict__ bias, float* __restrict__ out) {
    const int lane = threadIdx.x & 31;
    const int w    = threadIdx.x >> 5;
    const int wm   = w >> 2;            // 0..1
    const int wn   = w & 3;             // 0..3
    const int mt0  = blockIdx.y * 8  + wm * 4;   // 16-row tile index (of 512)
    const int nt0  = blockIdx.x * 16 + wn * 4;   // 16-col tile index (of 256)

    const v16bf* ap  = (const v16bf*)Ah;
    const v16bf* alp = (const v16bf*)Al;
    const v16bf* bp  = (const v16bf*)Bh;
    const v16bf* blp = (const v16bf*)Bl;

    v8f acc[4][4] = {};

    for (int kt = 0; kt < KTILES; ++kt) {
        v16bf a_h[4], b_h[4];
        v16bf a_l[4], b_l[4];
#pragma unroll
        for (int i = 0; i < 4; ++i) {
            int t = ((mt0 + i) * KTILES + kt) * 32 + lane;   // v16bf units
            a_h[i] = ap[t];
            if (SPLIT) a_l[i] = alp[t];
        }
#pragma unroll
        for (int j = 0; j < 4; ++j) {
            int t = ((nt0 + j) * KTILES + kt) * 32 + lane;
            b_h[j] = bp[t];
            if (SPLIT) b_l[j] = blp[t];
        }
#pragma unroll
        for (int i = 0; i < 4; ++i) {
#pragma unroll
            for (int j = 0; j < 4; ++j) {
                acc[i][j] = __builtin_amdgcn_wmma_f32_16x16x32_bf16(
                    false, a_h[i], false, b_h[j], (short)0, acc[i][j], false, false);
                if (SPLIT) {
                    acc[i][j] = __builtin_amdgcn_wmma_f32_16x16x32_bf16(
                        false, a_h[i], false, b_l[j], (short)0, acc[i][j], false, false);
                    acc[i][j] = __builtin_amdgcn_wmma_f32_16x16x32_bf16(
                        false, a_l[i], false, b_h[j], (short)0, acc[i][j], false, false);
                }
            }
        }
    }

    // Epilogue: C/D layout: VGPR r, lanes 0-15: (M=r, N=lane); lanes 16-31: (M=r+8, N=lane-16)
#pragma unroll
    for (int j = 0; j < 4; ++j) {
        int col = (nt0 + j) * 16 + (lane & 15);
        float bv = bias[col];
#pragma unroll
        for (int i = 0; i < 4; ++i) {
            int row0 = (mt0 + i) * 16 + ((lane >> 4) << 3);
#pragma unroll
            for (int r = 0; r < 8; ++r) {
                out[(size_t)(row0 + r) * N_DIM + col] = acc[i][j][r] + bv;
            }
        }
    }
}

// ---------------------------------------------------------------------------
// Fallback (tiny workspace): plain fp32 dot per output element.
// ---------------------------------------------------------------------------
__global__ void naive_kernel(const float* __restrict__ x, const float* __restrict__ W,
                             const float* __restrict__ b, float* __restrict__ out) {
    size_t idx = (size_t)blockIdx.x * 256 + threadIdx.x;
    int row = (int)(idx >> 12);
    int col = (int)(idx & 4095);
    float s = b[col];
    const float* xr = x + (size_t)row * K_DIM;
    const float* wr = W + (size_t)col * K_DIM;
    for (int k = 0; k < K_DIM; ++k) s += xr[k] * wr[k];
    out[idx] = s;
}

extern "C" void kernel_launch(void* const* d_in, const int* in_sizes, int n_in,
                              void* d_out, int out_size, void* d_ws, size_t ws_size,
                              hipStream_t stream) {
    const float* x = (const float*)d_in[0];
    const float* W = (const float*)d_in[1];
    const float* b = (const float*)d_in[2];
    float* out = (float*)d_out;

    const size_t TK = (size_t)T_DIM * K_DIM;   // x elems
    const size_t NK = (size_t)N_DIM * K_DIM;   // W elems
    const size_t need_split = (2 * TK + 2 * NK) * sizeof(__bf16);   // ~201 MB
    const size_t need_hi    = (TK + NK) * sizeof(__bf16);           // ~100 MB

    dim3 gemm_grid(N_DIM / 256, T_DIM / 128);   // 16 x 64
    dim3 block(256);

    if (ws_size >= need_split) {
        __bf16* Ah = (__bf16*)d_ws;
        __bf16* Al = Ah + TK;
        __bf16* Bh = Al + TK;
        __bf16* Bl = Bh + NK;
        pack_x_kernel<true><<<(unsigned)(TK / 256), block, 0, stream>>>(x, Ah, Al);
        pack_w_kernel<true><<<(unsigned)(NK / 256), block, 0, stream>>>(W, Bh, Bl);
        gemm_wmma_kernel<true><<<gemm_grid, block, 0, stream>>>(Ah, Al, Bh, Bl, b, out);
    } else if (ws_size >= need_hi) {
        __bf16* Ah = (__bf16*)d_ws;
        __bf16* Bh = Ah + TK;
        pack_x_kernel<false><<<(unsigned)(TK / 256), block, 0, stream>>>(x, Ah, nullptr);
        pack_w_kernel<false><<<(unsigned)(NK / 256), block, 0, stream>>>(W, Bh, nullptr);
        gemm_wmma_kernel<false><<<gemm_grid, block, 0, stream>>>(Ah, Ah, Bh, Bh, b, out);
    } else {
        size_t total = (size_t)T_DIM * N_DIM;
        naive_kernel<<<(unsigned)(total / 256), block, 0, stream>>>(x, W, b, out);
    }
}
